// LovaszSoftmaxLoss_79534204387675
// MI455X (gfx1250) — compile-verified, matrix operand order checked by reading, not run on hardware
//
#include <hip/hip_runtime.h>
#include <hip/hip_bf16.h>
#include <stdint.h>

// Lovasz-Softmax loss, MI455X (gfx1250).
// Pipeline per call:
//   k_init  : zero per-class fg counts
//   k_build : probas=exp(x), errors=|fg-p|, pack u64 key = (desc_float_key<<32)|fg,
//             one contiguous 2M-key plane per class; count gts per class.
//   8x (k_hist -> k_scanhist -> k_scatter): stable LSD radix sort, 4-bit digits
//             over the top 32 bits. Tiles staged to LDS with CDNA5 async loads.
//   k_loss  : per class, streaming scan of the sorted keys; WMMA f16 computes
//             16x32-tile prefix sums of fg; Jaccard grad evaluated locally:
//             contribution_i = e_i * (J(i,cf_i) - J(i-1, cf_i - fg_i)).
//   k_final : weighted mean over present classes.

#define PN 2000000
#define CN 19
#define TILE 2048
#define NT 977              // ceil(PN / TILE)
#define ITEMS 8

typedef __attribute__((ext_vector_type(16))) _Float16 v16h;
typedef __attribute__((ext_vector_type(8)))  float    v8f;
typedef unsigned int u32;
typedef unsigned long long u64;

__device__ __forceinline__ u32 enc_desc(float e) {
    u32 b = __float_as_uint(e);          // e >= 0
    return ~(b ^ 0x80000000u);           // ascending sort of this == descending e
}
__device__ __forceinline__ float dec_err(u32 k) {
    return __uint_as_float((~k) ^ 0x80000000u);
}

// ---------------------------------------------------------------- init
__global__ void k_init(u32* gts) {
    if (threadIdx.x < CN) gts[threadIdx.x] = 0u;
}

// ---------------------------------------------------------------- build keys
__global__ void k_build(const float* __restrict__ inp, const int* __restrict__ tgt,
                        u64* __restrict__ keys, u32* __restrict__ gts) {
    __shared__ u32 lg[CN];
    int t = threadIdx.x;
    if (t < CN) lg[t] = 0u;
    __syncthreads();
    int p = blockIdx.x * blockDim.x + t;
    if (p < PN) {
        int tc = tgt[p];
        atomicAdd(&lg[tc], 1u);
        const float* row = inp + (size_t)p * CN;
        #pragma unroll
        for (int c = 0; c < CN; ++c) {
            float prob = expf(row[c]);
            float fg = (tc == c) ? 1.0f : 0.0f;
            float err = fabsf(fg - prob);
            u64 key = ((u64)enc_desc(err) << 32) | (u64)((tc == c) ? 1u : 0u);
            keys[(size_t)c * PN + p] = key;
        }
    }
    __syncthreads();
    if (t < CN) atomicAdd(&gts[t], lg[t]);
}

// ---------------------------------------------------------------- histogram
__global__ void k_hist(const u64* __restrict__ src, u32* __restrict__ hist, int shift) {
    __shared__ __align__(16) u64 stage[TILE];
    __shared__ u32 cnt[16];
    int t = threadIdx.x;
    int tile = blockIdx.x, c = blockIdx.y;
    long long tileBase = (long long)tile * TILE;
    u32 lbase = (u32)(unsigned long long)(void*)&stage[0];
    // CDNA5 async copy: tile (16KB) -> LDS, 4 x b128 per lane
    #pragma unroll
    for (int i = 0; i < 4; ++i) {
        int e = i * 512 + 2 * t;
        long long g = tileBase + e; if (g > PN - 2) g = PN - 2;
        u32 lds  = lbase + (u32)e * 8u;
        u32 voff = (u32)(((size_t)c * PN + (size_t)g) * 8ull);
        asm volatile("global_load_async_to_lds_b128 %0, %1, %2"
                     :: "v"(lds), "v"(voff), "s"(src) : "memory");
    }
    if (t < 16) cnt[t] = 0u;
    asm volatile("s_wait_asynccnt 0" ::: "memory");
    __syncthreads();
    #pragma unroll
    for (int j = 0; j < ITEMS; ++j) {
        int idx = t * ITEMS + j;
        long long r = tileBase + idx;
        if (r < PN) {
            u32 d = (u32)((stage[idx] >> shift) & 15ull);
            atomicAdd(&cnt[d], 1u);
        }
    }
    __syncthreads();
    if (t < 16) hist[((size_t)c * 16 + t) * NT + tile] = cnt[t];
}

// ---------------------------------------------------------------- hist -> global bases
__global__ void k_scanhist(u32* __restrict__ hist) {
    __shared__ u32 red[256];
    __shared__ u32 totals[16];
    __shared__ u32 starts[16];
    int t = threadIdx.x, c = blockIdx.x;
    u32* hc = hist + (size_t)c * 16 * NT;
    for (int d = 0; d < 16; ++d) {
        u32 s = 0;
        for (int i = t; i < NT; i += 256) s += hc[d * NT + i];
        red[t] = s; __syncthreads();
        for (int off = 128; off > 0; off >>= 1) {
            if (t < off) red[t] += red[t + off];
            __syncthreads();
        }
        if (t == 0) totals[d] = red[0];
        __syncthreads();
    }
    if (t == 0) {
        u32 run = 0;
        for (int d = 0; d < 16; ++d) { starts[d] = run; run += totals[d]; }
    }
    __syncthreads();
    for (int d = 0; d < 16; ++d) {
        u32 base = starts[d];
        for (int cb = 0; cb < NT; cb += 256) {
            int i = cb + t;
            u32 v = (i < NT) ? hc[d * NT + i] : 0u;
            red[t] = v;
            for (int off = 1; off < 256; off <<= 1) {
                __syncthreads();
                u32 x = (t >= off) ? red[t - off] : 0u;
                __syncthreads();
                red[t] += x;
            }
            __syncthreads();
            u32 incl = red[t];
            u32 tot  = red[255];
            if (i < NT) hc[d * NT + i] = base + (incl - v);
            base += tot;
            __syncthreads();
        }
    }
}

// ---------------------------------------------------------------- stable scatter
__global__ void k_scatter(const u64* __restrict__ src, u64* __restrict__ dst,
                          const u32* __restrict__ hist, int shift) {
    __shared__ __align__(16) u64 stage[TILE];
    __shared__ u32 sc[8][256];
    __shared__ u32 hb[16];
    int t = threadIdx.x;
    int tile = blockIdx.x, c = blockIdx.y;
    long long tileBase = (long long)tile * TILE;
    u32 lbase = (u32)(unsigned long long)(void*)&stage[0];
    #pragma unroll
    for (int i = 0; i < 4; ++i) {
        int e = i * 512 + 2 * t;
        long long g = tileBase + e; if (g > PN - 2) g = PN - 2;
        u32 lds  = lbase + (u32)e * 8u;
        u32 voff = (u32)(((size_t)c * PN + (size_t)g) * 8ull);
        asm volatile("global_load_async_to_lds_b128 %0, %1, %2"
                     :: "v"(lds), "v"(voff), "s"(src) : "memory");
    }
    if (t < 16) hb[t] = hist[((size_t)c * 16 + t) * NT + tile];
    asm volatile("s_wait_asynccnt 0" ::: "memory");
    __syncthreads();

    u64 key[ITEMS]; u32 dig[ITEMS]; u32 rank[ITEMS]; int valid[ITEMS];
    u32 pc[8];
    #pragma unroll
    for (int k = 0; k < 8; ++k) pc[k] = 0u;
    // each thread owns ITEMS consecutive elements -> within-thread order == memory order
    #pragma unroll
    for (int j = 0; j < ITEMS; ++j) {
        int idx = t * ITEMS + j;
        long long r = tileBase + idx;
        valid[j] = (r < PN);
        key[j] = stage[idx];
        dig[j] = (u32)((key[j] >> shift) & 15ull);
        u32 wsel = dig[j] >> 1, hsel = (dig[j] & 1u) * 16u;
        rank[j] = (pc[wsel] >> hsel) & 0xffffu;
        if (valid[j]) pc[wsel] += (1u << hsel);
    }
    // inclusive scan across 256 threads of packed 16 x u16 digit counters
    #pragma unroll
    for (int k = 0; k < 8; ++k) sc[k][t] = pc[k];
    for (int off = 1; off < 256; off <<= 1) {
        __syncthreads();
        u32 tmp[8];
        #pragma unroll
        for (int k = 0; k < 8; ++k) tmp[k] = (t >= off) ? sc[k][t - off] : 0u;
        __syncthreads();
        #pragma unroll
        for (int k = 0; k < 8; ++k) sc[k][t] += tmp[k];
    }
    __syncthreads();
    u32 excl[8];
    #pragma unroll
    for (int k = 0; k < 8; ++k) excl[k] = sc[k][t] - pc[k];
    u64* dstC = dst + (size_t)c * PN;
    #pragma unroll
    for (int j = 0; j < ITEMS; ++j) {
        if (valid[j]) {
            u32 d = dig[j];
            u32 pos = hb[d] + ((excl[d >> 1] >> ((d & 1u) * 16u)) & 0xffffu) + rank[j];
            dstC[pos] = key[j];
        }
    }
}

// ---------------------------------------------------------------- Lovasz scan (WMMA)
__global__ void k_loss(const u64* __restrict__ keys, const u32* __restrict__ gts,
                       float* __restrict__ classLoss) {
    __shared__ float cumTile[8][16][16];
    __shared__ float waveTot[8];
    __shared__ double rbuf[256];
    int t = threadIdx.x, c = blockIdx.x;
    int w = t >> 5, lane = t & 31;
    int m = lane & 15;                     // row of A handled by this lane
    const u64* kc = keys + (size_t)c * PN;
    float gtsF = (float)gts[c];

    // K index owned by (lane, j) per the 16-bit A layout (16x32)
    int Kmap[16];
    #pragma unroll
    for (int j = 0; j < 16; ++j) {
        int K = (j < 8) ? j : j + 8;
        if (lane >= 16) K += 8;
        Kmap[j] = K;
    }
    // B[k][n] = (k <= 2n+1): WMMA yields inclusive fg-prefix at odd k per row
    v16h bmat = {};
    #pragma unroll
    for (int j = 0; j < 16; ++j)
        bmat[j] = (Kmap[j] <= 2 * m + 1) ? (_Float16)1.0f : (_Float16)0.0f;

    double acc = 0.0;
    float runBase = 0.0f;                  // fg count before current 4096-chunk
    const int ITER = (PN + 4095) / 4096;   // 8 waves x 512-elem tiles per iter
    for (int it = 0; it < ITER; ++it) {
        long long waveBase = (long long)it * 4096 + (long long)w * 512;
        v16h amat = {};
        float errv[16]; float fgv[16];
        #pragma unroll
        for (int j = 0; j < 16; j += 2) {  // (K,K+1) pairs -> aligned b128 loads
            long long r0 = waveBase + (long long)m * 32 + Kmap[j];
            long long rc = r0; if (rc > PN - 2) rc = PN - 2;
            ulonglong2 kv = *(const ulonglong2*)(kc + rc);
            float f0 = (float)(u32)(kv.x & 1ull), f1 = (float)(u32)(kv.y & 1ull);
            float e0 = dec_err((u32)(kv.x >> 32)), e1 = dec_err((u32)(kv.y >> 32));
            if (r0 >= PN)     { f0 = 0.0f; e0 = 0.0f; }
            if (r0 + 1 >= PN) { f1 = 0.0f; e1 = 0.0f; }
            fgv[j] = f0; fgv[j + 1] = f1; errv[j] = e0; errv[j + 1] = e1;
            amat[j]     = (_Float16)f0;
            amat[j + 1] = (_Float16)f1;
        }
        long long nb = waveBase + 4096 + (long long)m * 32;
        if (nb < PN) __builtin_prefetch(kc + nb, 0, 0);

        v8f cz = {};
        v8f d = __builtin_amdgcn_wmma_f32_16x16x32_f16(false, amat, false, bmat,
                                                       (short)0, cz, false, false);
        int mh = (lane >> 4) * 8;          // C/D layout: VGPR j -> row mh+j, col m
        #pragma unroll
        for (int j2 = 0; j2 < 8; ++j2)
            cumTile[w][mh + j2][m] = d[j2];
        __syncthreads();

        float tot = 0.0f, rowOff = 0.0f;   // exclusive row offsets within the wave tile
        #pragma unroll
        for (int m2 = 0; m2 < 16; ++m2) {
            if (m2 == m) rowOff = tot;
            tot += cumTile[w][m2][15];
        }
        if (lane == 0) waveTot[w] = tot;
        __syncthreads();
        float wavePre = 0.0f, grand = 0.0f;
        #pragma unroll
        for (int w2 = 0; w2 < 8; ++w2) {
            if (w2 == w) wavePre = grand;
            grand += waveTot[w2];
        }
        float base = runBase + wavePre + rowOff;
        #pragma unroll
        for (int j = 0; j < 16; ++j) {
            int K = Kmap[j];
            long long r = waveBase + (long long)m * 32 + K;
            if (r < PN) {
                float cl = (K & 1) ? cumTile[w][m][K >> 1]
                                   : (cumTile[w][m][K >> 1] - fgv[(j + 1) & 15]);
                float cumfg = base + cl;           // exact: integer < 2^24
                float fi = (float)(r + 1);
                float Jc = 1.0f - (gtsF - cumfg) / (gtsF + fi - cumfg);
                float Jp = 0.0f;
                if (r != 0) {
                    float cf2 = cumfg - fgv[j];
                    Jp = 1.0f - (gtsF - cf2) / (gtsF + (fi - 1.0f) - cf2);
                }
                acc += (double)(errv[j] * (Jc - Jp));
            }
        }
        runBase += grand;
        __syncthreads();
    }
    rbuf[t] = acc;
    __syncthreads();
    for (int off = 128; off > 0; off >>= 1) {
        if (t < off) rbuf[t] += rbuf[t + off];
        __syncthreads();
    }
    if (t == 0) classLoss[c] = (float)rbuf[0];
}

// ---------------------------------------------------------------- final mean
__global__ void k_final(const float* __restrict__ classLoss, const u32* __restrict__ gts,
                        float* __restrict__ out) {
    __shared__ float sl[32], sw[32];
    int t = threadIdx.x;
    float l = 0.0f, wv = 0.0f;
    if (t < CN) {
        wv = (gts[t] > 0u) ? 1.0f : 0.0f;
        l  = classLoss[t] * wv;
    }
    sl[t] = l; sw[t] = wv;
    __syncthreads();
    if (t == 0) {
        float a = 0.0f, b = 0.0f;
        for (int i = 0; i < CN; ++i) { a += sl[i]; b += sw[i]; }
        out[0] = a / fmaxf(b, 1.0f);
    }
}

// ---------------------------------------------------------------- launch
extern "C" void kernel_launch(void* const* d_in, const int* in_sizes, int n_in,
                              void* d_out, int out_size, void* d_ws, size_t ws_size,
                              hipStream_t stream) {
    (void)in_sizes; (void)n_in; (void)out_size; (void)ws_size;
    const float* inp = (const float*)d_in[0];
    const int*   tgt = (const int*)d_in[1];
    float* out = (float*)d_out;

    const size_t N = (size_t)PN * CN;
    u64* buf0 = (u64*)d_ws;
    u64* buf1 = buf0 + N;
    u32* hist = (u32*)(buf1 + N);
    u32* gts  = hist + (size_t)CN * 16 * NT;
    float* classLoss = (float*)(gts + CN);

    k_init<<<dim3(1), dim3(32), 0, stream>>>(gts);
    k_build<<<dim3((PN + 255) / 256), dim3(256), 0, stream>>>(inp, tgt, buf0, gts);
    for (int pass = 0; pass < 8; ++pass) {
        int shift = 32 + 4 * pass;
        u64* src = (pass & 1) ? buf1 : buf0;
        u64* dst = (pass & 1) ? buf0 : buf1;
        k_hist<<<dim3(NT, CN), dim3(256), 0, stream>>>(src, hist, shift);
        k_scanhist<<<dim3(CN), dim3(256), 0, stream>>>(hist);
        k_scatter<<<dim3(NT, CN), dim3(256), 0, stream>>>(src, dst, hist, shift);
    }
    // 8 passes -> sorted result back in buf0
    k_loss<<<dim3(CN), dim3(256), 0, stream>>>(buf0, gts, classLoss);
    k_final<<<dim3(1), dim3(32), 0, stream>>>(classLoss, gts, out);
}
